// TopicAttention_71786083385516
// MI455X (gfx1250) — matrix operand order repaired and checked by reading, not compiled, not run on hardware
//
#include <hip/hip_runtime.h>
#include <hip/hip_bf16.h>

#define Bn 128
#define Ln 65
#define Vn 50000
#define EMBn 300
#define Cn 128
#define Tn 11
#define CLSn 12
#define MAX_HOP 10
#define PAD_IDXn (Vn - 1)
#define BL (Bn * Ln)          // 8320 = 520*16 exactly
#define KCONV (Cn * 5)        // 640 = 20*32 exactly
#define KSTEPS (KCONV / 32)   // 20

typedef __attribute__((ext_vector_type(2)))  float  v2f;
typedef __attribute__((ext_vector_type(8)))  float  v8f;
typedef __attribute__((ext_vector_type(16))) __bf16 v16bf;

// ---------------- prep: max_len, pad mask, done flags ----------------
__global__ void k_prep(const int* __restrict__ lengths, int* __restrict__ pad,
                       int* __restrict__ done) {
  __shared__ int smax[128];
  int tid = threadIdx.x;
  smax[tid] = (tid < Bn) ? lengths[tid] : 0;
  __syncthreads();
  for (int s = 64; s > 0; s >>= 1) {
    if (tid < s) smax[tid] = max(smax[tid], smax[tid + s]);
    __syncthreads();
  }
  int maxlen = smax[0];
  for (int i = tid; i < BL; i += 128) {
    int b = i / Ln, l = i % Ln;
    int len = lengths[b];
    int eff = (len == maxlen) ? Ln : len;
    pad[i] = (l >= eff) ? 1 : 0;
  }
  if (tid < Tn) done[tid] = 0;
}

// ---------------- embedding + reduction GEMM: f32 WMMA 16x16x4 ----------------
// x0[r, n] = sum_k (emb[idx_r, k] + pm * pos[l, k]) * W_red[k, n] + b_red[n]
__global__ void k_embed(const int* __restrict__ x_idx, const float* __restrict__ emb,
                        const float* __restrict__ pos, const float* __restrict__ W_red,
                        const float* __restrict__ b_red, const int* __restrict__ pad,
                        float* __restrict__ x0) {
  int wave = threadIdx.x >> 5;
  int lane = threadIdx.x & 31;
  int tile = blockIdx.x * 8 + wave;   // 4160 tiles total (520 * 8)
  int tm = tile >> 3;                 // 0..519
  int tn = tile & 7;                  // 0..7
  int mrow = lane & 15;
  int khalf = lane >> 4;              // 0: K0,K1  1: K2,K3
  int r = tm * 16 + mrow;             // row (b,l) flat, < 8320 always
  int b = r / Ln, l = r % Ln;
  int idx = x_idx[b * Ln + l];
  const float* erow = emb + (long)idx * EMBn;
  const float* prow = pos + (long)l * EMBn;
  float pm = (idx != PAD_IDXn) ? 1.f : 0.f;   // branch-free: EXEC stays all-ones
  int n = tn * 16 + mrow;

  v8f acc = {};
  for (int k0 = 0; k0 < EMBn; k0 += 4) {
    int ka = k0 + 2 * khalf;
    v2f a, bb;
    a.x = erow[ka]     + pm * prow[ka];
    a.y = erow[ka + 1] + pm * prow[ka + 1];
    bb.x = W_red[ka * Cn + n];
    bb.y = W_red[(ka + 1) * Cn + n];
    acc = __builtin_amdgcn_wmma_f32_16x16x4_f32(false, a, false, bb,
                                                (short)0, acc, false, false);
  }
  float brd = b_red[n];
#pragma unroll
  for (int vg = 0; vg < 8; vg++) {
    int m = tm * 16 + khalf * 8 + vg;        // C/D: lanes16-31 hold M=8..15
    float val = acc[vg] + brd;
    if (pad[m]) val = 0.f;
    x0[(long)m * Cn + n] = val;
  }
}

// ---------------- broadcast per-topic state (f32 + bf16 shadow) ----------------
__global__ void k_init_state(const float* __restrict__ x0, const float* __restrict__ queries,
                             float* __restrict__ xs, __bf16* __restrict__ xbf,
                             float* __restrict__ qs) {
  long i = (long)blockIdx.x * blockDim.x + threadIdx.x;
  const long nx = (long)Tn * BL * Cn;
  if (i < nx) {
    float v = x0[i % ((long)BL * Cn)];
    xs[i] = v;
    xbf[i] = (__bf16)v;
  }
  const long nq = (long)Tn * Bn * Cn;
  if (i < nq) {
    int t = (int)(i / (Bn * Cn));
    int c = (int)(i % Cn);
    qs[i] = queries[t * Cn + c];
  }
}

// ---------------- conv weights: im2col repack, TRANSPOSED bf16 ----------------
// wbfT[t][co][kc],  kc = ci*5 + kk  <->  conv_w[t][co][ci][kk]
__global__ void k_wconv(const float* __restrict__ conv_w, __bf16* __restrict__ wbfT) {
  int i = blockIdx.x * blockDim.x + threadIdx.x;
  if (i >= Tn * Cn * KCONV) return;
  int kc = i % KCONV;
  int co = (i / KCONV) % Cn;
  int t  = i / (KCONV * Cn);
  int ci = kc / 5, kk = kc % 5;
  wbfT[i] = (__bf16)conv_w[((t * Cn + co) * Cn + ci) * 5 + kk];
}

// ---------------- attention energies + masked softmax ----------------
__global__ void k_energy(const float* __restrict__ xs, const float* __restrict__ qs,
                         const int* __restrict__ pad, float* __restrict__ probs) {
  int tb = blockIdx.x;            // t*B + b
  int b = tb % Bn;
  __shared__ float sq[Cn];
  __shared__ float e[Ln];
  __shared__ float rinv;
  int tid = threadIdx.x;          // 128
  sq[tid] = qs[(long)tb * Cn + tid];
  __syncthreads();
  if (tid < Ln) {
    const float* xr = xs + (long)tb * Ln * Cn + (long)tid * Cn;
    float s = 0.f;
    for (int c = 0; c < Cn; c++) s += xr[c] * sq[c];
    e[tid] = pad[b * Ln + tid] ? -__builtin_inff() : s;
  }
  __syncthreads();
  if (tid == 0) {
    float m = -__builtin_inff();
    for (int l = 0; l < Ln; l++) m = fmaxf(m, e[l]);
    float s = 0.f;
    for (int l = 0; l < Ln; l++) {
      float p = pad[b * Ln + l] ? 0.f : expf(e[l] - m);
      e[l] = p; s += p;
    }
    rinv = 1.f / s;
  }
  __syncthreads();
  if (tid < Ln) probs[(long)tb * Ln + tid] = e[tid] * rinv;
}

// ---------------- conv1d as GEMM: bf16 WMMA 16x16x32, LDS-staged A ----------------
// Block = 256 thr (8 waves) = one M-tile x all 8 N-tiles; A tile im2col'd once
// into LDS in exact per-lane fragment order (32B contiguous per lane per kstep).
__global__ void k_conv(const __bf16* __restrict__ xbf, const __bf16* __restrict__ wbfT,
                       const float* __restrict__ conv_b, float* __restrict__ h) {
  __shared__ alignas(32) __bf16 As[KSTEPS * 32 * 16];   // 20 KB
  int tid = threadIdx.x;
  int t  = blockIdx.x / 520;
  int tm = blockIdx.x % 520;

  // ---- cooperative im2col fill: As[((kstep*32)+lane)*16 + i] ----
  for (int e = tid; e < KSTEPS * 32 * 16; e += 256) {
    int i = e & 15;
    int ln = (e >> 4) & 31;
    int kstep = e >> 9;
    int hi = ln >> 4;
    int kc = kstep * 32 + (hi ? 16 : 0) + i;
    int ci = kc / 5;
    int kk = kc - ci * 5;
    int r = tm * 16 + (ln & 15);
    int l = r % Ln;
    int posl = l + kk - 2;
    __bf16 v = (__bf16)0.f;
    if (posl >= 0 && posl < Ln)
      v = xbf[((long)t * BL + (r - l) + posl) * Cn + ci];
    As[e] = v;
  }
  __syncthreads();

  // ---- compute: wave tn handles N-tile tn ----
  int tn = tid >> 5;
  int lane = tid & 31;
  int mrow = lane & 15;
  int hi = lane >> 4;
  int co = tn * 16 + mrow;
  const __bf16* wbT = wbfT + ((long)t * Cn + co) * KCONV;   // contiguous K per lane
  int bofs = hi ? 16 : 0;   // B, contiguous k like A: lanes0-15 K0-15, lanes16-31 K16-31

  v8f acc = {};
  for (int kstep = 0; kstep < KSTEPS; kstep++) {
    v16bf a  = *reinterpret_cast<const v16bf*>(&As[((kstep * 32) + lane) * 16]);
    v16bf bm = *reinterpret_cast<const v16bf*>(wbT + kstep * 32 + bofs);
    acc = __builtin_amdgcn_wmma_f32_16x16x32_bf16(false, a, false, bm,
                                                  (short)0, acc, false, false);
  }
  float bias = conv_b[t * Cn + co];
#pragma unroll
  for (int vg = 0; vg < 8; vg++) {
    int m = tm * 16 + hi * 8 + vg;
    h[((long)t * BL + m) * Cn + co] = acc[vg] + bias;
  }
}

// ---------------- batchnorm stats, two-pass ----------------
__global__ void k_bnstat1(const float* __restrict__ h, float* __restrict__ psum,
                          float* __restrict__ psq) {
  int t = blockIdx.x / 65, chunk = blockIdx.x % 65;   // 8320 = 65*128
  int c = threadIdx.x;
  const float* hp = h + ((long)t * BL + chunk * 128) * Cn + c;
  float s = 0.f, q = 0.f;
  for (int i = 0; i < 128; i++) { float v = hp[(long)i * Cn]; s += v; q += v * v; }
  psum[(long)blockIdx.x * Cn + c] = s;
  psq[(long)blockIdx.x * Cn + c] = q;
}

__global__ void k_bnstat2(const float* __restrict__ psum, const float* __restrict__ psq,
                          float* __restrict__ mean, float* __restrict__ rstd) {
  int t = blockIdx.x, c = threadIdx.x;
  float s = 0.f, q = 0.f;
  for (int ch = 0; ch < 65; ch++) {
    s += psum[((long)t * 65 + ch) * Cn + c];
    q += psq[((long)t * 65 + ch) * Cn + c];
  }
  float m = s / (float)BL;
  float v = q / (float)BL - m * m;
  mean[t * Cn + c] = m;
  rstd[t * Cn + c] = rsqrtf(v + 1e-5f);
}

// ---------------- BN + leaky + pool + halt head + state update ----------------
__global__ void k_update(const float* __restrict__ h, const float* __restrict__ probs,
                         const int* __restrict__ pad, const float* __restrict__ mean,
                         const float* __restrict__ rstd, const float* __restrict__ bn_g,
                         const float* __restrict__ bn_b, const float* __restrict__ halt_W,
                         const float* __restrict__ halt_b, const int* __restrict__ done,
                         float* __restrict__ xs, __bf16* __restrict__ xbf,
                         float* __restrict__ qs, int* __restrict__ halt) {
  int tb = blockIdx.x;
  int t = tb / Bn, b = tb % Bn;
  int c = threadIdx.x;                 // 128
  int dn = done[t];
  float mn = mean[t * Cn + c], rs = rstd[t * Cn + c];
  float g = bn_g[t * Cn + c], be = bn_b[t * Cn + c];
  const float* hp = h + (long)tb * Ln * Cn + c;
  const float* pr = probs + (long)tb * Ln;
  float*  xp  = xs  + (long)tb * Ln * Cn + c;
  __bf16* xbp = xbf + (long)tb * Ln * Cn + c;
  float o = 0.f;
  for (int l = 0; l < Ln; l++) {
    float v = hp[(long)l * Cn];
    float xn = (v - mn) * rs * g + be;
    xn = (xn > 0.f) ? xn : 0.01f * xn;
    if (pad[b * Ln + l]) xn = 0.f;
    o += xn * pr[l];
    if (!dn) { xp[(long)l * Cn] = xn; xbp[(long)l * Cn] = (__bf16)xn; }
  }
  float q = qs[(long)tb * Cn + c];
  float oq = o + q;
  __shared__ float s0[128], s1[128];
  s0[c] = oq * halt_W[(t * Cn + c) * 2 + 0];
  s1[c] = oq * halt_W[(t * Cn + c) * 2 + 1];
  __syncthreads();
  for (int s = 64; s > 0; s >>= 1) {
    if (c < s) { s0[c] += s0[c + s]; s1[c] += s1[c + s]; }
    __syncthreads();
  }
  if (c == 0) {
    float l0 = s0[0] + halt_b[t * 2 + 0];
    float l1 = s1[0] + halt_b[t * 2 + 1];
    halt[tb] = (l1 > l0) ? 1 : 0;
  }
  qs[(long)tb * Cn + c] = dn ? q : oq;
}

__global__ void k_done(const int* __restrict__ halt, int* __restrict__ done) {
  int t = blockIdx.x;
  __shared__ int any;
  if (threadIdx.x == 0) any = 0;
  __syncthreads();
  if (threadIdx.x < Bn && halt[t * Bn + threadIdx.x]) any = 1;
  __syncthreads();
  if (threadIdx.x == 0 && any) done[t] = 1;
}

// ---------------- output head + sigmoid ----------------
__global__ void k_out(const float* __restrict__ qs, const float* __restrict__ W_out,
                      const float* __restrict__ b_out, float* __restrict__ out) {
  int b = blockIdx.x;
  int j = threadIdx.x;
  if (j >= CLSn) return;
  float s = b_out[j];
  for (int t = 0; t < Tn; t++)
    for (int c = 0; c < Cn; c++)
      s += qs[((long)t * Bn + b) * Cn + c] * W_out[(t * Cn + c) * CLSn + j];
  out[b * CLSn + j] = 1.f / (1.f + expf(-s));
}

// ---------------- query orthogonality regularizer ----------------
__global__ void k_reg(const float* __restrict__ queries, float* __restrict__ outreg) {
  __shared__ float nrm[Tn];
  __shared__ float red[128];
  __shared__ float acc;
  int c = threadIdx.x;
  for (int t = 0; t < Tn; t++) {
    float v = queries[t * Cn + c];
    red[c] = v * v;
    __syncthreads();
    for (int s = 64; s > 0; s >>= 1) { if (c < s) red[c] += red[c + s]; __syncthreads(); }
    if (c == 0) nrm[t] = sqrtf(red[0]);
    __syncthreads();
  }
  if (c == 0) acc = 0.f;
  __syncthreads();
  for (int i = 0; i < Tn; i++)
    for (int j = 0; j < Tn; j++) {
      red[c] = (queries[i * Cn + c] / nrm[i]) * (queries[j * Cn + c] / nrm[j]);
      __syncthreads();
      for (int s = 64; s > 0; s >>= 1) { if (c < s) red[c] += red[c + s]; __syncthreads(); }
      if (c == 0) { float g = red[0] - (i == j ? 1.f : 0.f); acc += g * g; }
      __syncthreads();
    }
  if (c == 0) outreg[0] = sqrtf(acc);
}

extern "C" void kernel_launch(void* const* d_in, const int* in_sizes, int n_in,
                              void* d_out, int out_size, void* d_ws, size_t ws_size,
                              hipStream_t stream) {
  const int*   x_indices = (const int*)d_in[0];
  const int*   lengths   = (const int*)d_in[1];
  const float* emb       = (const float*)d_in[2];
  const float* pos       = (const float*)d_in[3];
  const float* W_red     = (const float*)d_in[4];
  const float* b_red     = (const float*)d_in[5];
  const float* queries   = (const float*)d_in[6];
  const float* conv_w    = (const float*)d_in[7];
  const float* conv_b    = (const float*)d_in[8];
  const float* bn_g      = (const float*)d_in[9];
  const float* bn_b      = (const float*)d_in[10];
  const float* halt_W    = (const float*)d_in[11];
  const float* halt_b    = (const float*)d_in[12];
  const float* W_out     = (const float*)d_in[13];
  const float* b_out     = (const float*)d_in[14];
  float* out = (float*)d_out;

  char* ws = (char*)d_ws;
  size_t off = 0;
  auto carve = [&](size_t bytes) -> void* {
    void* p = ws + off;
    off = (off + bytes + 255) & ~(size_t)255;
    return p;
  };
  int*    pad   = (int*)   carve((size_t)BL * 4);
  int*    done  = (int*)   carve((size_t)Tn * 4);
  int*    halt  = (int*)   carve((size_t)Tn * Bn * 4);
  float*  x0    = (float*) carve((size_t)BL * Cn * 4);
  float*  xs    = (float*) carve((size_t)Tn * BL * Cn * 4);
  __bf16* xbf   = (__bf16*)carve((size_t)Tn * BL * Cn * 2);
  float*  qs    = (float*) carve((size_t)Tn * Bn * Cn * 4);
  float*  probs = (float*) carve((size_t)Tn * BL * 4);
  float*  hbuf  = (float*) carve((size_t)Tn * BL * Cn * 4);
  __bf16* wbfT  = (__bf16*)carve((size_t)Tn * Cn * KCONV * 2);
  float*  psum  = (float*) carve((size_t)Tn * 65 * Cn * 4);
  float*  psq   = (float*) carve((size_t)Tn * 65 * Cn * 4);
  float*  meanb = (float*) carve((size_t)Tn * Cn * 4);
  float*  rstdb = (float*) carve((size_t)Tn * Cn * 4);

  k_prep<<<1, 128, 0, stream>>>(lengths, pad, done);
  k_embed<<<520, 256, 0, stream>>>(x_indices, emb, pos, W_red, b_red, pad, x0);
  {
    long nx = (long)Tn * BL * Cn;
    int blocks = (int)((nx + 255) / 256);
    k_init_state<<<blocks, 256, 0, stream>>>(x0, queries, xs, xbf, qs);
  }
  {
    int nw = Tn * Cn * KCONV;
    k_wconv<<<(nw + 255) / 256, 256, 0, stream>>>(conv_w, wbfT);
  }
  for (int hop = 0; hop < MAX_HOP; hop++) {
    k_energy<<<Tn * Bn, 128, 0, stream>>>(xs, qs, pad, probs);
    k_conv<<<Tn * 520, 256, 0, stream>>>(xbf, wbfT, conv_b, hbuf);
    k_bnstat1<<<Tn * 65, 128, 0, stream>>>(hbuf, psum, psq);
    k_bnstat2<<<Tn, 128, 0, stream>>>(psum, psq, meanb, rstdb);
    k_update<<<Tn * Bn, 128, 0, stream>>>(hbuf, probs, pad, meanb, rstdb, bn_g, bn_b,
                                          halt_W, halt_b, done, xs, xbf, qs, halt);
    k_done<<<Tn, 128, 0, stream>>>(halt, done);
  }
  k_out<<<Bn, 32, 0, stream>>>(qs, W_out, b_out, out);
  k_reg<<<1, 128, 0, stream>>>(queries, out + Bn * CLSn);
}
